// BertSelfAttention_70102456205778
// MI455X (gfx1250) — matrix operand order repaired and checked by reading, not compiled, not run on hardware
//
#include <hip/hip_runtime.h>

// ---- WMMA fragment types (wave32, gfx1250) ----
typedef __attribute__((ext_vector_type(16))) _Float16 v16h;
typedef __attribute__((ext_vector_type(8)))  _Float16 v8h;
typedef __attribute__((ext_vector_type(8)))  float    v8f;

union HFrag { v16h v; v8h h[2]; };

#define WMMA_F16(a, b, c) \
    __builtin_amdgcn_wmma_f32_16x16x32_f16(false, (a), false, (b), (short)0, (c), false, false)

// Async copy 16B/lane from global to LDS (ASYNCcnt-tracked, §08 async tensor doc).
// LDS operand = flat pointer truncated to 32 bits.
#define ASYNC_G2L_B128(ldsaddr_u32, gptr)                                  \
    asm volatile("global_load_async_to_lds_b128 %0, %1, off"               \
                 :: "v"(ldsaddr_u32), "v"(gptr) : "memory")

#define WAIT_ASYNC(n) asm volatile("s_wait_asynccnt " #n ::: "memory")

// Problem constants: B=4, S=2048, D=1024, H=16, HD=64, BH=64, M=B*S=8192

// =====================================================================
// Kernel 0: one-shot fp32 -> f16 convert (makes operands raw-copyable
// so the hot loops can use async LDS DMA with no cvt / VGPR round-trip)
// =====================================================================
__global__ __launch_bounds__(256) void cvt_f16_kernel(
    const float* __restrict__ src, _Float16* __restrict__ dst, int n8)
{
    const int i = blockIdx.x * 256 + threadIdx.x;
    if (i < n8) {
        const float4* p = reinterpret_cast<const float4*>(src) + (size_t)i * 2;
        float4 a = p[0], b = p[1];
        v8h h = { (_Float16)a.x,(_Float16)a.y,(_Float16)a.z,(_Float16)a.w,
                  (_Float16)b.x,(_Float16)b.y,(_Float16)b.z,(_Float16)b.w };
        *(reinterpret_cast<v8h*>(dst) + i) = h;
    }
}

// =====================================================================
// Kernel 1: fused QKV projection.  y = X @ W^T + b  (W stored [out,in])
// blockIdx.z selects {Q,K,V}.  64x64 tile / 128-thread block.  LDS is
// double-buffered and filled by GLOBAL_LOAD_ASYNC_TO_LDS_B128.
// Q,K stored f16 as [BH][S][64]; V stored f16 TRANSPOSED as [BH][64][S].
// =====================================================================
__global__ __launch_bounds__(128) void qkv_gemm_kernel(
    const _Float16* __restrict__ Xh,
    const _Float16* __restrict__ wqh, const float* __restrict__ bq,
    const _Float16* __restrict__ wkh, const float* __restrict__ bk,
    const _Float16* __restrict__ wvh, const float* __restrict__ bv,
    _Float16* __restrict__ qh, _Float16* __restrict__ kh, _Float16* __restrict__ vt)
{
    const int z = blockIdx.z;
    const _Float16* W    = (z == 0) ? wqh : (z == 1) ? wkh : wvh;
    const float*    bias = (z == 0) ? bq  : (z == 1) ? bk  : bv;
    _Float16*       out  = (z == 0) ? qh  : (z == 1) ? kh  : vt;

    const int tid  = threadIdx.x;
    const int wave = tid >> 5;
    const int lane = tid & 31;
    const int lh   = lane >> 4;
    const int ln   = lane & 15;

    const int mbase = blockIdx.x * 64;
    const int nbase = blockIdx.y * 64;

    // padded stride 40 halves (20 dwords) -> conflict-free b128 frag reads
    __shared__ __align__(16) _Float16 Xs[2][64][40];
    __shared__ __align__(16) _Float16 Ws[2][64][40];

    v8f acc[4] = {};

    // staging: each thread owns one 16B chunk per matrix per K-step
    const int lrow = tid >> 1;          // 0..63
    const int lseg = (tid & 1) * 16;    // element offset 0 or 16

    const _Float16* gx = Xh + (size_t)(mbase + lrow) * 1024 + lseg;
    const _Float16* gw = W  + (size_t)(nbase + lrow) * 1024 + lseg;
    const unsigned  lx[2] = { (unsigned)(size_t)&Xs[0][lrow][lseg],
                              (unsigned)(size_t)&Xs[1][lrow][lseg] };
    const unsigned  lw[2] = { (unsigned)(size_t)&Ws[0][lrow][lseg],
                              (unsigned)(size_t)&Ws[1][lrow][lseg] };

    // prologue: fill buffer 0
    ASYNC_G2L_B128(lx[0], gx);
    ASYNC_G2L_B128(lw[0], gw);

    for (int kk = 0; kk < 1024; kk += 32) {
        const int buf = (kk >> 5) & 1;
        if (kk + 32 < 1024) {           // fill next buffer, retire current
            ASYNC_G2L_B128(lx[buf ^ 1], gx + kk + 32);
            ASYNC_G2L_B128(lw[buf ^ 1], gw + kk + 32);
            WAIT_ASYNC(0x2);
        } else {
            WAIT_ASYNC(0x0);
        }
        __syncthreads();

        // A frag 16x32: lanes 0-15 -> K0-7 & K16-23 ; lanes 16-31 -> K8-15 & K24-31
        HFrag a;
        a.h[0] = *(const v8h*)&Xs[buf][wave * 16 + ln][lh * 8];
        a.h[1] = *(const v8h*)&Xs[buf][wave * 16 + ln][16 + lh * 8];
#pragma unroll
        for (int t = 0; t < 4; ++t) {
            // B frag 32x16: lanes 0-15 -> K0-7 & K16-23 ; lanes 16-31 -> K8-15 & K24-31
            HFrag b;
            b.h[0] = *(const v8h*)&Ws[buf][t * 16 + ln][lh * 8];
            b.h[1] = *(const v8h*)&Ws[buf][t * 16 + ln][16 + lh * 8];
            acc[t] = WMMA_F16(a.v, b.v, acc[t]);
        }
        __syncthreads();   // all waves done reading buf before it is refilled
    }

    // ---- epilogue: +bias, f16 store in attention-friendly layout ----
#pragma unroll
    for (int t = 0; t < 4; ++t) {
        const int n = nbase + t * 16 + ln;
        const float bval = bias[n];
        const int hh = n >> 6, dd = n & 63;
#pragma unroll
        for (int r = 0; r < 8; ++r) {
            const int m = mbase + wave * 16 + r + lh * 8;  // C layout: rows r / r+8
            const int bb = m >> 11, ss = m & 2047;
            const float val = acc[t][r] + bval;
            size_t idx = (z < 2)
                ? ((size_t)((bb * 16 + hh) * 2048 + ss)) * 64 + dd   // [BH][S][64]
                : ((size_t)((bb * 16 + hh) * 64 + dd)) * 2048 + ss;  // [BH][64][S]
            out[idx] = (_Float16)val;
        }
    }
}

// =====================================================================
// Kernel 2: flash attention.  One (b,h) x 64-query tile per block; each
// of 4 waves owns 16 query rows.  K/V chunks async-staged to double-
// buffered LDS (shared by all 4 waves).  Online softmax, pad-mask bias.
// =====================================================================
__global__ __launch_bounds__(128) void attn_kernel(
    const _Float16* __restrict__ qh, const _Float16* __restrict__ kh,
    const _Float16* __restrict__ vt, const float* __restrict__ pad,
    float* __restrict__ out)
{
    const int tid  = threadIdx.x;
    const int wave = tid >> 5;
    const int lane = tid & 31;
    const int lh   = lane >> 4;
    const int ln   = lane & 15;

    const int bh = blockIdx.y;      // 0..63
    const int b  = bh >> 4;
    const int h  = bh & 15;
    const int qbase = blockIdx.x * 64 + wave * 16;

    // stride 72 halves (36 dwords): 16 distinct banks for b128 frag reads
    __shared__ __align__(16) _Float16 Ks[2][64][72];
    __shared__ __align__(16) _Float16 Vs[2][64][72];
    __shared__ __align__(16) _Float16 Ps[4][16][72];

    // ---- staging geometry: 64x64 f16 chunk = 512 x 16B, 4 chunks/thread ----
    int srow[4], sseg[4];
    unsigned lK[2][4], lV[2][4];
#pragma unroll
    for (int c = 0; c < 4; ++c) {
        const int idx = c * 128 + tid;
        srow[c] = idx >> 3;
        sseg[c] = (idx & 7) * 8;
        lK[0][c] = (unsigned)(size_t)&Ks[0][srow[c]][sseg[c]];
        lK[1][c] = (unsigned)(size_t)&Ks[1][srow[c]][sseg[c]];
        lV[0][c] = (unsigned)(size_t)&Vs[0][srow[c]][sseg[c]];
        lV[1][c] = (unsigned)(size_t)&Vs[1][srow[c]][sseg[c]];
    }
    // prologue: fill buffer 0 with keys [0,64)
#pragma unroll
    for (int c = 0; c < 4; ++c) {
        const _Float16* pk = kh + ((size_t)bh * 2048 + srow[c]) * 64 + sseg[c];
        const _Float16* pv = vt + ((size_t)bh * 64 + srow[c]) * 2048 + sseg[c];
        ASYNC_G2L_B128(lK[0][c], pk);
        ASYNC_G2L_B128(lV[0][c], pv);
    }

    // ---- Q fragments (A-layout), pre-scaled by 1/sqrt(64) ----
    HFrag aq[2];
    {
        const _Float16* qrow = qh + ((size_t)bh * 2048 + (qbase + ln)) * 64;
#pragma unroll
        for (int j = 0; j < 2; ++j) {
            aq[j].h[0] = *(const v8h*)(qrow + j * 32 + lh * 8);
            aq[j].h[1] = *(const v8h*)(qrow + j * 32 + 16 + lh * 8);
#pragma unroll
            for (int e = 0; e < 16; ++e) aq[j].v[e] *= (_Float16)0.125f;
        }
    }

    float mrow[8], lrow[8];
    v8f o[4] = {};
#pragma unroll
    for (int r = 0; r < 8; ++r) { mrow[r] = -3.0e38f; lrow[r] = 0.0f; }

    for (int kt = 0; kt < 2048; kt += 64) {
        const int buf = (kt >> 6) & 1;
        if (kt + 64 < 2048) {           // fill next buffer, retire current
#pragma unroll
            for (int c = 0; c < 4; ++c) {
                const _Float16* pk = kh + ((size_t)bh * 2048 + kt + 64 + srow[c]) * 64 + sseg[c];
                const _Float16* pv = vt + ((size_t)bh * 64 + srow[c]) * 2048 + kt + 64 + sseg[c];
                ASYNC_G2L_B128(lK[buf ^ 1][c], pk);
                ASYNC_G2L_B128(lV[buf ^ 1][c], pv);
            }
            WAIT_ASYNC(0x8);
        } else {
            WAIT_ASYNC(0x0);
        }
        __syncthreads();

        // ---- S = (Q/sqrt(d)) K^T over this 64-key chunk ----
        float biasv[4];
        v8f s[4] = {};
#pragma unroll
        for (int t = 0; t < 4; ++t) {
            const int key = kt + t * 16 + ln;
            biasv[t] = (pad[b * 2048 + key] - 1.0f) * 1e9f;  // (1-pad)*-1e9
#pragma unroll
            for (int j = 0; j < 2; ++j) {
                HFrag bk;
                bk.h[0] = *(const v8h*)&Ks[buf][t * 16 + ln][j * 32 + lh * 8];
                bk.h[1] = *(const v8h*)&Ks[buf][t * 16 + ln][j * 32 + 16 + lh * 8];
                s[t] = WMMA_F16(aq[j].v, bk.v, s[t]);
            }
        }

        // ---- online softmax (row r of each half lives in VGPR r) ----
#pragma unroll
        for (int r = 0; r < 8; ++r) {
            s[0][r] += biasv[0]; s[1][r] += biasv[1];
            s[2][r] += biasv[2]; s[3][r] += biasv[3];
            float mv = fmaxf(fmaxf(s[0][r], s[1][r]), fmaxf(s[2][r], s[3][r]));
            mv = fmaxf(mv, __shfl_xor(mv, 1, 32));
            mv = fmaxf(mv, __shfl_xor(mv, 2, 32));
            mv = fmaxf(mv, __shfl_xor(mv, 4, 32));
            mv = fmaxf(mv, __shfl_xor(mv, 8, 32));
            const float mn   = fmaxf(mrow[r], mv);
            const float corr = __expf(mrow[r] - mn);
            mrow[r] = mn;
            lrow[r] *= corr;
            o[0][r] *= corr; o[1][r] *= corr; o[2][r] *= corr; o[3][r] *= corr;
            const float p0 = __expf(s[0][r] - mn);
            const float p1 = __expf(s[1][r] - mn);
            const float p2 = __expf(s[2][r] - mn);
            const float p3 = __expf(s[3][r] - mn);
            s[0][r] = p0; s[1][r] = p1; s[2][r] = p2; s[3][r] = p3;
            float rs = p0 + p1 + p2 + p3;
            rs += __shfl_xor(rs, 1, 32);
            rs += __shfl_xor(rs, 2, 32);
            rs += __shfl_xor(rs, 4, 32);
            rs += __shfl_xor(rs, 8, 32);
            lrow[r] += rs;
        }

        // ---- P: C-layout regs -> LDS f16 (A-layout reload) ----
#pragma unroll
        for (int t = 0; t < 4; ++t)
#pragma unroll
            for (int r = 0; r < 8; ++r)
                Ps[wave][r + lh * 8][t * 16 + ln] = (_Float16)s[t][r];
        __syncthreads();

        HFrag ap[2];
#pragma unroll
        for (int j = 0; j < 2; ++j) {
            ap[j].h[0] = *(const v8h*)&Ps[wave][ln][j * 32 + lh * 8];
            ap[j].h[1] = *(const v8h*)&Ps[wave][ln][j * 32 + 16 + lh * 8];
        }

        // ---- O += P V  (V pre-transposed: rows are d, cols are keys) ----
#pragma unroll
        for (int t = 0; t < 4; ++t) {
#pragma unroll
            for (int j = 0; j < 2; ++j) {
                HFrag bv;
                bv.h[0] = *(const v8h*)&Vs[buf][t * 16 + ln][j * 32 + lh * 8];
                bv.h[1] = *(const v8h*)&Vs[buf][t * 16 + ln][j * 32 + 16 + lh * 8];
                o[t] = WMMA_F16(ap[j].v, bv.v, o[t]);
            }
        }
        __syncthreads();   // Ps + Ks/Vs[buf] reads done before refill/rewrite
    }

    // ---- normalize and store fp32 context [B,S,H*HD] ----
#pragma unroll
    for (int t = 0; t < 4; ++t)
#pragma unroll
        for (int r = 0; r < 8; ++r) {
            const int q = qbase + r + lh * 8;
            const int d = t * 16 + ln;
            out[((size_t)(b * 2048 + q)) * 1024 + h * 64 + d] = o[t][r] / lrow[r];
        }
}

// =====================================================================
extern "C" void kernel_launch(void* const* d_in, const int* in_sizes, int n_in,
                              void* d_out, int out_size, void* d_ws, size_t ws_size,
                              hipStream_t stream) {
    const float* hs  = (const float*)d_in[0];
    const float* pad = (const float*)d_in[1];
    const float* wq  = (const float*)d_in[2];
    const float* bq  = (const float*)d_in[3];
    const float* wk  = (const float*)d_in[4];
    const float* bk  = (const float*)d_in[5];
    const float* wv  = (const float*)d_in[6];
    const float* bv  = (const float*)d_in[7];

    const size_t eBSD = (size_t)8 * 1024 * 1024;   // B*S*D = 8388608
    const size_t eDD  = (size_t)1024 * 1024;       // D*D

    _Float16* qh  = (_Float16*)d_ws;
    _Float16* kh  = qh  + eBSD;
    _Float16* vt  = kh  + eBSD;
    _Float16* xh  = vt  + eBSD;
    _Float16* wqh = xh  + eBSD;
    _Float16* wkh = wqh + eDD;
    _Float16* wvh = wkh + eDD;

    // one-shot fp32 -> f16 conversions (raw-copyable operands for async DMA)
    cvt_f16_kernel<<<dim3((unsigned)(eBSD / 8 / 256)), dim3(256), 0, stream>>>(hs, xh, (int)(eBSD / 8));
    cvt_f16_kernel<<<dim3((unsigned)(eDD  / 8 / 256)), dim3(256), 0, stream>>>(wq, wqh, (int)(eDD / 8));
    cvt_f16_kernel<<<dim3((unsigned)(eDD  / 8 / 256)), dim3(256), 0, stream>>>(wk, wkh, (int)(eDD / 8));
    cvt_f16_kernel<<<dim3((unsigned)(eDD  / 8 / 256)), dim3(256), 0, stream>>>(wv, wvh, (int)(eDD / 8));

    // M/64 = 128 tiles, N/64 = 16 tiles, z = {Q,K,V}
    qkv_gemm_kernel<<<dim3(128, 16, 3), dim3(128), 0, stream>>>(
        xh, wqh, bq, wkh, bk, wvh, bv, qh, kh, vt);

    // 32 query tiles x 64 (b,h) pairs
    attn_kernel<<<dim3(32, 64), dim3(128), 0, stream>>>(
        qh, kh, vt, pad, (float*)d_out);
}
